// S6Block_63041529971052
// MI455X (gfx1250) — matrix operand order repaired and checked by reading, not compiled, not run on hardware
//
#include <hip/hip_runtime.h>
#include <hip/hip_bf16.h>
#include <math.h>
#include <stdint.h>

typedef __bf16 bf16_t;
typedef __attribute__((ext_vector_type(16))) __bf16 v16bf;
typedef __attribute__((ext_vector_type(8)))  float  v8f;
typedef __attribute__((ext_vector_type(4)))  float  f32x4;
typedef __attribute__((ext_vector_type(4)))  unsigned int u32x4;
typedef __attribute__((ext_vector_type(8)))  int    i32x8;
typedef __attribute__((ext_vector_type(4)))  int    i32x4;

#define BATCH 2
#define LSEQ  2048
#define DMODEL 1024
#define NST   16
#define RDT   64
#define MTOK  (BATCH * LSEQ)   // 4096 tokens

// ---------------- workspace layout (bytes) ----------------
static constexpr size_t align256(size_t v) { return (v + 255) & ~(size_t)255; }
static constexpr size_t XB_OFF   = 0;
static constexpr size_t XB_SZ    = (size_t)MTOK * DMODEL * 2;        // bf16 x
static constexpr size_t WINB_OFF = align256(XB_OFF + XB_SZ);
static constexpr size_t WINB_SZ  = (size_t)DMODEL * 32 * 2;          // packed W_in
static constexpr size_t W1B_OFF  = align256(WINB_OFF + WINB_SZ);
static constexpr size_t W1B_SZ   = (size_t)DMODEL * RDT * 2;         // packed W_dt1
static constexpr size_t W2B_OFF  = align256(W1B_OFF + W1B_SZ);
static constexpr size_t W2B_SZ   = (size_t)RDT * DMODEL * 2;         // packed W_dt2
static constexpr size_t BC_OFF   = align256(W2B_OFF + W2B_SZ);
static constexpr size_t BC_SZ    = (size_t)MTOK * 32 * 4;            // Bm|Cm f32
static constexpr size_t DT_OFF   = align256(BC_OFF + BC_SZ);
static constexpr size_t DT_SZ    = (size_t)MTOK * DMODEL * 4;        // f32 dt

// ---------------- kernel 0a: f32 -> bf16 (activations) ----------------
__global__ void cvt_bf16_kernel(const float* __restrict__ src,
                                bf16_t* __restrict__ dst, int n) {
  int i = blockIdx.x * blockDim.x + threadIdx.x;
  if (i < n) dst[i] = (bf16_t)src[i];
}

// ---------------- kernel 0b: pack weight into WMMA B-fragment order ----
// Packed layout: [(k0/32)*(N/16) + nt][lane][16 elems], each lane's 16 bf16
// contiguous (32 B) so the proj kernel loads a fragment with two b128 loads.
// B fragment map used here (contiguous k, same as A): lanes 0-15 hold K k0..k0+15, lanes 16-31
// hold K k0+16..k0+31; column N = nt*16 + (lane&15); element j -> K = kb + j.
__global__ void pack_b_kernel(const float* __restrict__ W,
                              bf16_t* __restrict__ Wp, int K, int Ncols) {
  int idx = blockIdx.x * 256 + threadIdx.x;
  int total = (K >> 5) * (Ncols >> 4) * 32 * 16;
  if (idx >= total) return;
  int j    = idx & 15;
  int lane = (idx >> 4) & 31;
  int rest = idx >> 9;
  int ntiles = Ncols >> 4;
  int nt = rest % ntiles;
  int k0 = rest / ntiles;
  int kb  = k0 * 32 + ((lane >> 4) ? 16 : 0) + j;
  int col = nt * 16 + (lane & 15);
  Wp[idx] = (bf16_t)W[(size_t)kb * Ncols + col];
}

__device__ __forceinline__ void load_bfrag(const bf16_t* __restrict__ Wp,
                                           int kchunk, int ntiles, int nt,
                                           int lane, v16bf* dst) {
  union { v16bf v; f32x4 q[2]; } bw;
  const f32x4* p = (const f32x4*)(Wp +
      ((((size_t)kchunk * ntiles + nt) * 32 + lane) << 4));
  bw.q[0] = p[0];
  bw.q[1] = p[1];
  *dst = bw.v;
}

// ---------------- kernel 1: projections via WMMA ----------------
// One wave (32 threads) per 16-token tile:
//   BC = x @ W_in + b_in                       (16 x 32)
//   T  = x @ W_dt1 + b_dt1                     (16 x 64, staged in LDS)
//   dt = clip(softplus(T @ W_dt2 + b_dt2))     (16 x 1024)
__global__ void proj_kernel(const bf16_t* __restrict__ xb,
                            const bf16_t* __restrict__ WinP,
                            const bf16_t* __restrict__ W1P,
                            const bf16_t* __restrict__ W2P,
                            const float*  __restrict__ b_in,
                            const float*  __restrict__ b_dt1,
                            const float*  __restrict__ b_dt2,
                            float* __restrict__ BC,
                            float* __restrict__ dt) {
  __shared__ float Tl[16 * RDT];

  const int lane = threadIdx.x & 31;
  const int half = lane >> 4;
  const int l15  = lane & 15;
  const int m0   = blockIdx.x * 16;

  v8f accY[2]; v8f accT[4];
#pragma unroll
  for (int i = 0; i < 2; ++i) accY[i] = {};
#pragma unroll
  for (int i = 0; i < 4; ++i) accT[i] = {};

  const bf16_t* xrow = xb + (size_t)(m0 + l15) * DMODEL;

  for (int k0 = 0; k0 < DMODEL; k0 += 32) {
    if (k0 + 32 < DMODEL) __builtin_prefetch(xrow + k0 + 32, 0, 1);

    // A fragment of x tile (16x32 bf16), contiguous k order, same as packed B:
    //   lanes 0-15: K = k0+{0..15}; lanes 16-31: K = k0+{16..31}
    union { v16bf v; f32x4 q[2]; bf16_t e[16]; } af;
    const bf16_t* ab = xrow + k0 + (half ? 16 : 0);
#pragma unroll
    for (int j = 0; j < 8; ++j) { af.e[j] = ab[j]; af.e[8 + j] = ab[8 + j]; }

    const int kchunk = k0 >> 5;
#pragma unroll
    for (int nt = 0; nt < 2; ++nt) {
      v16bf bw; load_bfrag(WinP, kchunk, 2, nt, lane, &bw);
      accY[nt] = __builtin_amdgcn_wmma_f32_16x16x32_bf16(
          false, af.v, false, bw, (short)0, accY[nt], false, false);
    }
#pragma unroll
    for (int nt = 0; nt < 4; ++nt) {
      v16bf bw; load_bfrag(W1P, kchunk, 4, nt, lane, &bw);
      accT[nt] = __builtin_amdgcn_wmma_f32_16x16x32_bf16(
          false, af.v, false, bw, (short)0, accT[nt], false, false);
    }
  }

  // BC = Y + b_in. C/D layout: element r -> row M = r + 8*half, col = lane&15.
#pragma unroll
  for (int nt = 0; nt < 2; ++nt) {
    union { v8f v; float e[8]; } c; c.v = accY[nt];
#pragma unroll
    for (int r = 0; r < 8; ++r) {
      int M = r + 8 * half, col = nt * 16 + l15;
      BC[(size_t)(m0 + M) * 32 + col] = c.e[r] + b_in[col];
    }
  }
#pragma unroll
  for (int nt = 0; nt < 4; ++nt) {
    union { v8f v; float e[8]; } c; c.v = accT[nt];
#pragma unroll
    for (int r = 0; r < 8; ++r) {
      int M = r + 8 * half, col = nt * 16 + l15;
      Tl[M * RDT + col] = c.e[r] + b_dt1[col];
    }
  }
  __syncthreads();

  // A fragments of T (two 16x32 K-chunks) from LDS.
  union { v16bf v; bf16_t e[16]; } a3[2];
#pragma unroll
  for (int kc = 0; kc < 2; ++kc) {
    const int kbA = kc * 32 + (half ? 16 : 0);
#pragma unroll
    for (int j = 0; j < 8; ++j) {
      a3[kc].e[j]     = (bf16_t)Tl[l15 * RDT + kbA + j];
      a3[kc].e[8 + j] = (bf16_t)Tl[l15 * RDT + kbA + 8 + j];
    }
  }

  // dt_raw = T @ W_dt2 (K=64): 64 output tiles x 2 K-chunks.
  for (int nt = 0; nt < 64; ++nt) {
    v8f acc = {};
#pragma unroll
    for (int kc = 0; kc < 2; ++kc) {
      v16bf bw; load_bfrag(W2P, kc, 64, nt, lane, &bw);
      acc = __builtin_amdgcn_wmma_f32_16x16x32_bf16(
          false, a3[kc].v, false, bw, (short)0, acc, false, false);
    }
    union { v8f v; float e[8]; } c; c.v = acc;
#pragma unroll
    for (int r = 0; r < 8; ++r) {
      int M = r + 8 * half, col = nt * 16 + l15;
      float z = c.e[r] + b_dt2[col];
      float sp = (z > 20.0f) ? z : log1pf(__expf(z));   // softplus
      sp = fminf(fmaxf(sp, 1e-7f), 1e6f);
      dt[(size_t)(m0 + M) * DMODEL + col] = sp;
    }
  }
}

// ---------------- TDM: issue one 2D f32 tile load into LDS ----------------
// D# per ISA ch.8: group0 = {count=1, lds_addr, global_addr(57b), type=2},
// group1 = {data_size=4B, tensor_dim = tile dims (no OOB), tile dims,
//           tensor_dim0_stride = row stride}. 2D -> groups 2/3 zero.
__device__ __forceinline__ void tdm_load_2d(uint32_t lds_off,
                                            const float* gptr,
                                            uint32_t tile_w, uint32_t tile_h,
                                            uint32_t stride_elems) {
  uint64_t ga = (uint64_t)(uintptr_t)gptr;
  u32x4 g0 = { 1u,                               // count = 1 valid descriptor
               lds_off,                          // LDS byte address
               (uint32_t)ga,                     // global addr [31:0]
               (uint32_t)((ga >> 32) & 0x01FFFFFFu) | (2u << 30) }; // [56:32]|type=2
  i32x8 g1 = { (int)0x00020000u,                 // data_size = 2 (4 bytes)
               (int)((tile_w & 0xFFFFu) << 16),  // tensor_dim0[15:0]
               (int)((tile_w >> 16) | ((tile_h & 0xFFFFu) << 16)), // td0 hi | td1 lo
               (int)((tile_h >> 16) | ((tile_w & 0xFFFFu) << 16)), // td1 hi | tile_dim0
               (int)(tile_h & 0xFFFFu),          // tile_dim1 | tile_dim2=0
               (int)stride_elems,                // tensor_dim0_stride[31:0]
               0, 0 };
  i32x4 z4 = {0, 0, 0, 0};
#if __has_include(<hip/amd_detail/amd_gfx1250_TDM.h>)
  i32x8 z8 = {0, 0, 0, 0, 0, 0, 0, 0};
  __builtin_amdgcn_tensor_load_to_lds(g0, g1, z4, z4, z8, 0);
#else
  __builtin_amdgcn_tensor_load_to_lds(g0, g1, z4, z4, 0);
#endif
}

// ---------------- kernel 2: selective scan ----------------
// Block = 256 threads = 16 channels x 16 states; grid = BATCH * (DMODEL/16).
// Chunks of 64 timesteps staged into LDS by the Tensor Data Mover, double
// buffered: wave 0 issues chunk c+1 (3 TDM loads) then s_wait_tensorcnt(3)
// to know chunk c landed (TDM ops from one wave complete in order).
// Recurrence: h = exp(A*dt)*h + ((exp(A*dt)-1)/A)*Bm*x  (expm1 folded, A const)
// out[t,d] = sum_n h*Cm (xor-shuffles within each 16-lane half) + D*x.
__global__ void scan_kernel(const float* __restrict__ x,
                            const float* __restrict__ dt,
                            const float* __restrict__ BC,
                            const float* __restrict__ A,
                            const float* __restrict__ Dv,
                            float* __restrict__ out) {
  constexpr int CH = 64;
  constexpr int NCHUNK = LSEQ / CH;
  __shared__ float xs[2][CH * 16];
  __shared__ float dts[2][CH * 16];
  __shared__ float bcs[2][CH * 32];

  const int tid = threadIdx.x;
  const int n  = tid & 15;        // state index
  const int dl = tid >> 4;        // local channel 0..15
  const int b  = blockIdx.x >> 6;
  const int d0 = (blockIdx.x & 63) * 16;
  const int d  = d0 + dl;

  const float An   = A[(size_t)d * NST + n];   // = -(1+n)
  const float invA = 1.0f / An;
  const float Dd   = Dv[d];
  const size_t tokBase = (size_t)b * LSEQ;

  const bool issuer = (tid < 32);   // wave 0 drives the TDM

  auto issue_chunk = [&](int c, int buf) {
    const size_t t0 = (size_t)c * CH;
    tdm_load_2d((uint32_t)(uintptr_t)(void*)&xs[buf][0],
                x + (tokBase + t0) * DMODEL + d0, 16, CH, DMODEL);
    tdm_load_2d((uint32_t)(uintptr_t)(void*)&dts[buf][0],
                dt + (tokBase + t0) * DMODEL + d0, 16, CH, DMODEL);
    tdm_load_2d((uint32_t)(uintptr_t)(void*)&bcs[buf][0],
                BC + (tokBase + t0) * 32, 32, CH, 32);
  };

  if (issuer) issue_chunk(0, 0);

  float h = 0.0f;
  for (int c = 0; c < NCHUNK; ++c) {
    const int cur = c & 1;
    if (issuer) {
      if (c + 1 < NCHUNK) {
        issue_chunk(c + 1, cur ^ 1);
        __builtin_amdgcn_s_wait_tensorcnt(3);   // chunk c complete
      } else {
        __builtin_amdgcn_s_wait_tensorcnt(0);
      }
    }
    __syncthreads();   // chunk c visible to all waves

    const float* xc = xs[cur];
    const float* dc = dts[cur];
    const float* bc = bcs[cur];
    const int t0 = c * CH;
    for (int i = 0; i < CH; ++i) {
      float xv  = xc[i * 16 + dl];
      float dtv = dc[i * 16 + dl];
      float bm  = bc[i * 32 + n];
      float cm  = bc[i * 32 + 16 + n];
      float a   = __expf(An * dtv);
      h = a * h + (a - 1.0f) * invA * bm * xv;
      float contrib = h * cm;
      contrib += __shfl_xor(contrib, 1, 32);
      contrib += __shfl_xor(contrib, 2, 32);
      contrib += __shfl_xor(contrib, 4, 32);
      contrib += __shfl_xor(contrib, 8, 32);
      if (n == 0)
        out[(tokBase + t0 + i) * DMODEL + d] = contrib + Dd * xv;
    }
    __syncthreads();   // reads of this buffer done before TDM refills it
  }
}

// ---------------- launcher ----------------
extern "C" void kernel_launch(void* const* d_in, const int* in_sizes, int n_in,
                              void* d_out, int out_size, void* d_ws, size_t ws_size,
                              hipStream_t stream) {
  const float* x     = (const float*)d_in[0];
  const float* W_in  = (const float*)d_in[1];
  const float* b_in  = (const float*)d_in[2];
  const float* W_dt1 = (const float*)d_in[3];
  const float* b_dt1 = (const float*)d_in[4];
  const float* W_dt2 = (const float*)d_in[5];
  const float* b_dt2 = (const float*)d_in[6];
  const float* A     = (const float*)d_in[7];
  const float* Dv    = (const float*)d_in[8];
  float* out = (float*)d_out;

  char* ws = (char*)d_ws;
  bf16_t* xb   = (bf16_t*)(ws + XB_OFF);
  bf16_t* WinP = (bf16_t*)(ws + WINB_OFF);
  bf16_t* W1P  = (bf16_t*)(ws + W1B_OFF);
  bf16_t* W2P  = (bf16_t*)(ws + W2B_OFF);
  float*  BC   = (float*)(ws + BC_OFF);
  float*  dt   = (float*)(ws + DT_OFF);

  const int NX = MTOK * DMODEL;
  cvt_bf16_kernel<<<(NX + 255) / 256, 256, 0, stream>>>(x, xb, NX);
  pack_b_kernel<<<(DMODEL * 32 + 255) / 256, 256, 0, stream>>>(W_in, WinP, DMODEL, 32);
  pack_b_kernel<<<(DMODEL * RDT + 255) / 256, 256, 0, stream>>>(W_dt1, W1P, DMODEL, RDT);
  pack_b_kernel<<<(RDT * DMODEL + 255) / 256, 256, 0, stream>>>(W_dt2, W2P, RDT, DMODEL);

  proj_kernel<<<MTOK / 16, 32, 0, stream>>>(xb, WinP, W1P, W2P,
                                            b_in, b_dt1, b_dt2, BC, dt);

  scan_kernel<<<BATCH * (DMODEL / 16), 256, 0, stream>>>(x, dt, BC, A, Dv, out);
}
